// Contrast_71090298683407
// MI455X (gfx1250) — compile-verified, matrix operand order checked by reading, not run on hardware
//
#include <hip/hip_runtime.h>

// ---------------- problem constants ----------------
#define TWO_N   8192          // 2 * BATCH
#define HALF_N  4096
#define DIM     2048
#define INV_T   2.0f          // 1 / TEMP (TEMP = 0.5)

// ---------------- tiling ----------------
#define BM      64            // rows per workgroup
#define BN      128           // cols per column-iteration
#define KC      64            // K halfs per staged chunk (2 WMMA k-steps)
#define NKC     (DIM / KC)    // 32
#define LPA     (KC + 8)      // LDS pitch in halfs (16B pad -> conflict-free)
#define CSPLIT  4             // column split across gridDim.y
#define CITERS  ((TWO_N / CSPLIT) / BN)  // 16

typedef __attribute__((ext_vector_type(16))) _Float16 v16h;
typedef __attribute__((ext_vector_type(8)))  _Float16 v8h;
typedef __attribute__((ext_vector_type(4)))  _Float16 v4h;
typedef __attribute__((ext_vector_type(8)))  float    v8f;
typedef __attribute__((ext_vector_type(4)))  int      v4i;

#define SHUF16(lo, hi) \
  __builtin_shufflevector(lo, hi, 0,1,2,3,4,5,6,7,8,9,10,11,12,13,14,15)

// ---- async global->LDS path (CDNA5), guarded so this compiles either way ----
#if __has_builtin(__builtin_amdgcn_global_load_async_to_lds_b128)
#define USE_ASYNC 1
#else
#define USE_ASYNC 0
#endif

__device__ __forceinline__ void lds_cp16(_Float16* lp, const _Float16* gp) {
#if USE_ASYNC
  __builtin_amdgcn_global_load_async_to_lds_b128(
      (__attribute__((address_space(1))) v4i*)gp,
      (__attribute__((address_space(3))) v4i*)lp, 0, 0);
#else
  *(uint4*)lp = *(const uint4*)gp;   // global_load_b128 + ds_store_b128
#endif
}

#if USE_ASYNC
#if __has_builtin(__builtin_amdgcn_s_wait_asynccnt)
#define AWAIT(n) __builtin_amdgcn_s_wait_asynccnt(n)
#else
#define AWAIT(n) asm volatile("s_wait_asynccnt %0" ::"i"(n) : "memory")
#endif
#else
#define AWAIT(n)
#endif

// ============ Kernel 1: L2-normalize rows, emit f16 z (8192 x 2048) ============
__global__ __launch_bounds__(256) void nrm_kernel(const float* __restrict__ xi,
                                                  const float* __restrict__ xj,
                                                  _Float16* __restrict__ z) {
  const int row  = (blockIdx.x * blockDim.x + threadIdx.x) >> 5;  // wave per row
  const int lane = threadIdx.x & 31;
  const float* src = (row < HALF_N) ? (xi + (size_t)row * DIM)
                                    : (xj + (size_t)(row - HALF_N) * DIM);
  const float4* s4 = (const float4*)src;
  float4 v[16];
  float ss = 0.f;
#pragma unroll
  for (int j = 0; j < 16; ++j) {
    v[j] = s4[lane + 32 * j];
    ss += v[j].x * v[j].x + v[j].y * v[j].y + v[j].z * v[j].z + v[j].w * v[j].w;
  }
#pragma unroll
  for (int off = 16; off >= 1; off >>= 1) ss += __shfl_xor(ss, off, 32);
  const float sc = 1.0f / fmaxf(sqrtf(ss), 1e-12f);
  _Float16* zr = z + (size_t)row * DIM;
#pragma unroll
  for (int j = 0; j < 16; ++j) {
    v4h h;
    h.x = (_Float16)(v[j].x * sc);
    h.y = (_Float16)(v[j].y * sc);
    h.z = (_Float16)(v[j].z * sc);
    h.w = (_Float16)(v[j].w * sc);
    *(v4h*)(zr + (size_t)(lane + 32 * j) * 4) = h;
  }
}

// ============ Kernel 2: fused sim = z z^T tile GEMM + exp row-sums ============
__global__ __launch_bounds__(256) void simloss_kernel(
    const _Float16* __restrict__ z, float* __restrict__ rowSumG,
    float* __restrict__ posG, float* __restrict__ diagG) {
  __shared__ __align__(16) _Float16 As[2][BM * LPA];
  __shared__ __align__(16) _Float16 Bs[2][BN * LPA];
  __shared__ float rowSumL[BM];

  const int tid      = threadIdx.x;
  const int lane     = tid & 31;
  const int wave     = tid >> 5;          // 8 waves: 4 row-groups x 2 col-groups
  const int lm       = lane & 15;         // row/col within a 16x16 tile
  const int hi       = lane >> 4;         // upper-half-lane flag
  const int kh       = hi << 3;           // K sub-offset 0 or 8
  const int waveRowL = (wave & 3) * 16;   // 0,16,32,48
  const int waveColL = (wave >> 2) * 64;  // 0 or 64
  const int wgRow    = blockIdx.x * BM;
  const int colGroup = blockIdx.y * (TWO_N / CSPLIT);

  if (tid < BM) rowSumL[tid] = 0.f;

  // stage one KC-chunk of A (64 rows) and B (128 "cols" = z rows) into buf
  auto stage = [&](int buf, int kc, int colBase) {
    const _Float16* gA = z + (size_t)wgRow * DIM + kc * KC;
#pragma unroll
    for (int j = 0; j < 2; ++j) {               // 512 x 16B for A
      const int i = tid + j * 256;
      const int r = i >> 3, c8 = (i & 7) << 3;
      lds_cp16(&As[buf][r * LPA + c8], gA + (size_t)r * DIM + c8);
    }
    const _Float16* gB = z + (size_t)colBase * DIM + kc * KC;
#pragma unroll
    for (int j = 0; j < 4; ++j) {               // 1024 x 16B for B
      const int i = tid + j * 256;
      const int r = i >> 3, c8 = (i & 7) << 3;
      lds_cp16(&Bs[buf][r * LPA + c8], gB + (size_t)r * DIM + c8);
    }
  };

  float rAcc[8] = {0.f, 0.f, 0.f, 0.f, 0.f, 0.f, 0.f, 0.f};

  for (int ci = 0; ci < CITERS; ++ci) {
    const int colBase = colGroup + ci * BN;
    v8f acc[4];
#pragma unroll
    for (int t = 0; t < 4; ++t) acc[t] = (v8f){};

    stage(0, 0, colBase);
    for (int kc = 0; kc < NKC; ++kc) {
      const int cur = kc & 1;
      if (kc + 1 < NKC) {         // prefetch next chunk into the other buffer
        stage(cur ^ 1, kc + 1, colBase);
        AWAIT(6);                 // allow the 6 just-issued per-thread loads
      } else {
        AWAIT(0);
      }
      __syncthreads();

      const _Float16* Ab = As[cur];
      const _Float16* Bb = Bs[cur];
#pragma unroll
      for (int ks = 0; ks < KC; ks += 32) {
        const _Float16* ap = Ab + (waveRowL + lm) * LPA + ks + kh;
        const v16h a = SHUF16(*(const v8h*)ap, *(const v8h*)(ap + 16));
#pragma unroll
        for (int t = 0; t < 4; ++t) {
          const _Float16* bp = Bb + (waveColL + t * 16 + lm) * LPA + ks + kh;
          const v16h b = SHUF16(*(const v8h*)bp, *(const v8h*)(bp + 16));
          acc[t] = __builtin_amdgcn_wmma_f32_16x16x32_f16(
              false, a, false, b, (short)0, acc[t], false, false);
        }
      }
      __syncthreads();
    }

    // epilogue: exp, row-sum accumulate, capture diagonal and positive
    const int rB = wgRow + waveRowL + 8 * hi;   // this lane's row base
#pragma unroll
    for (int t = 0; t < 4; ++t) {
      const int cg = colBase + waveColL + t * 16 + lm;  // this lane's column
#pragma unroll
      for (int v = 0; v < 8; ++v) {
        const float s = acc[t][v];
        const float e = __expf(s * INV_T);
        rAcc[v] += e;
        const int r = rB + v;
        if (cg == r)            diagG[r] = e;   // unique writer
        if (cg == (r ^ HALF_N)) posG[r]  = s;   // unique writer
      }
    }
  }

#pragma unroll
  for (int v = 0; v < 8; ++v)
    atomicAdd(&rowSumL[waveRowL + 8 * hi + v], rAcc[v]);   // ds_add_f32
  __syncthreads();
  if (tid < BM) atomicAdd(&rowSumG[wgRow + tid], rowSumL[tid]);
}

// ============ Kernel 3: per-row loss + global reduction ============
__global__ __launch_bounds__(256) void fin_kernel(const float* __restrict__ rowSum,
                                                  const float* __restrict__ pos,
                                                  const float* __restrict__ diag,
                                                  float* __restrict__ out) {
  const int r = blockIdx.x * 256 + threadIdx.x;
  float v = -INV_T * pos[r] + __logf(rowSum[r] - diag[r]);
#pragma unroll
  for (int off = 16; off >= 1; off >>= 1) v += __shfl_xor(v, off, 32);
  __shared__ float wsum[8];
  if ((threadIdx.x & 31) == 0) wsum[threadIdx.x >> 5] = v;
  __syncthreads();
  if (threadIdx.x == 0) {
    float t = 0.f;
#pragma unroll
    for (int i = 0; i < 8; ++i) t += wsum[i];
    atomicAdd(out, t * (1.0f / TWO_N));
  }
}

// ============ launch ============
extern "C" void kernel_launch(void* const* d_in, const int* in_sizes, int n_in,
                              void* d_out, int out_size, void* d_ws, size_t ws_size,
                              hipStream_t stream) {
  const float* xi = (const float*)d_in[0];
  const float* xj = (const float*)d_in[1];
  float* out = (float*)d_out;

  char* ws = (char*)d_ws;
  _Float16* z = (_Float16*)ws;                       // 32 MB f16 z
  const size_t zBytes = (size_t)TWO_N * DIM * sizeof(_Float16);
  float* rowSum = (float*)(ws + zBytes);             // 8192 f32
  float* pos    = rowSum + TWO_N;                    // 8192 f32
  float* diag   = pos + TWO_N;                       // 8192 f32

  (void)hipMemsetAsync(rowSum, 0, TWO_N * sizeof(float), stream);
  (void)hipMemsetAsync(out, 0, sizeof(float), stream);

  nrm_kernel<<<(TWO_N * 32) / 256, 256, 0, stream>>>(xi, xj, z);
  simloss_kernel<<<dim3(TWO_N / BM, CSPLIT), 256, 0, stream>>>(z, rowSum, pos, diag);
  fin_kernel<<<TWO_N / 256, 256, 0, stream>>>(rowSum, pos, diag, out);
}
